// RGCNModel_50079318671420
// MI455X (gfx1250) — matrix operand order repaired and checked by reading, not compiled, not run on hardware
//
#include <hip/hip_runtime.h>

typedef __attribute__((ext_vector_type(16))) __bf16 v16bf;
typedef __attribute__((ext_vector_type(8)))  __bf16 v8bf;
typedef __attribute__((ext_vector_type(8)))  float  v8f;

static constexpr int N_NEWS = 10000;
static constexpr int N_INT  = 50000;
static constexpr int NN     = 60000;     // total nodes
static constexpr int F_IN   = 768;
static constexpr int HD     = 256;
static constexpr int KCAT   = 1024;      // [x | agg0 | agg1 | agg2]
static constexpr int OUTD   = 8;

// ---------------- utility kernels ----------------

__global__ void k_zero(float* __restrict__ p, long n) {
  long i = (long)blockIdx.x * blockDim.x + threadIdx.x;
  if (i < n) p[i] = 0.f;
}

__global__ void k_cvt(const float* __restrict__ s, __bf16* __restrict__ d, long n) {
  long i = (long)blockIdx.x * blockDim.x + threadIdx.x;
  if (i < n) d[i] = (__bf16)s[i];
}

// Transpose+convert W[K x 256] (row-major f32) into Bt[n*ldb + koff + k] (bf16)
__global__ void k_packBt(const float* __restrict__ W, int K,
                         __bf16* __restrict__ Bt, int ldb, int koff) {
  long i = (long)blockIdx.x * blockDim.x + threadIdx.x;
  long tot = (long)K * HD;
  if (i < tot) {
    int k = (int)(i / HD);
    int n = (int)(i % HD);
    Bt[(long)n * ldb + koff + k] = (__bf16)W[i];
  }
}

__global__ void k_deg(const int* __restrict__ dst, int nE, int doff, float* __restrict__ deg) {
  int e = blockIdx.x * blockDim.x + threadIdx.x;
  if (e < nE) atomicAdd(&deg[dst[e] + doff], 1.0f);
}

__global__ void k_inv(float* __restrict__ p, long n) {
  long i = (long)blockIdx.x * blockDim.x + threadIdx.x;
  if (i < n) { float v = p[i]; p[i] = (v > 0.f) ? (1.0f / v) : 0.f; }
}

// mean-aggregation scatter: agg[dst] += x[src]  (x is bf16 rows of Abuf, stride KCAT)
__global__ void k_scatter(const int* __restrict__ src, const int* __restrict__ dst,
                          int nE, int soff, int doff,
                          const __bf16* __restrict__ X, float* __restrict__ agg) {
  long t = (long)blockIdx.x * blockDim.x + threadIdx.x;
  long tot = (long)nE * 64;                 // 64 chunks of 4 floats = 256 feats
  if (t < tot) {
    int e = (int)(t >> 6);
    int c = (int)(t & 63) * 4;
    long s = (long)src[e] + soff;
    long d = (long)dst[e] + doff;
    const __bf16* xp = X + s * KCAT + c;
    float* ap = agg + d * HD + c;
    atomicAdd(ap + 0, (float)xp[0]);
    atomicAdd(ap + 1, (float)xp[1]);
    atomicAdd(ap + 2, (float)xp[2]);
    atomicAdd(ap + 3, (float)xp[3]);
  }
}

// Abuf[n, coloff + h] = bf16(agg[n,h] * inv[n])
__global__ void k_scale_cvt(const float* __restrict__ agg, const float* __restrict__ inv,
                            __bf16* __restrict__ Abuf, int coloff) {
  long t = (long)blockIdx.x * blockDim.x + threadIdx.x;
  long tot = (long)NN * HD;
  if (t < tot) {
    long n = t >> 8;
    int  h = (int)(t & 255);
    Abuf[n * KCAT + coloff + h] = (__bf16)(agg[t] * inv[n]);
  }
}

// ---------------- WMMA GEMM ----------------
// C[M x 256] = A[M x K] @ Bt^T (+bias, optional relu), bf16 in / fp32 acc / bf16 out.
// Bt is packed [256 x K] row-major (i.e. W transposed). Block = 4 waves; each wave
// does a 16x64 slab (4 WMMA tiles); grid.x = M/16. In-place safe via __syncthreads.
__global__ __launch_bounds__(128) void k_gemm(
    const __bf16* __restrict__ A, int lda,
    const __bf16* __restrict__ Bt, int K,
    const float* __restrict__ bias,
    __bf16* __restrict__ C, int ldc, int relu) {
  const int row0 = blockIdx.x * 16;
  const int wave = threadIdx.x >> 5;
  const int lane = threadIdx.x & 31;
  const int m    = lane & 15;
  const int hv   = lane >> 4;
  const int col0 = wave * 64;

  v8f acc[4] = {};

  // A fragment: lane covers 16 contiguous K in [16*hv,16*hv+16)
  const __bf16* Arow = A + (long)(row0 + m) * lda + 16 * hv;
  // B fragment: lane n holds 16 contiguous K values of column (col0+nt*16+m)
  const __bf16* B0 = Bt + (long)(col0 + m) * K + 16 * hv;

  for (int k0 = 0; k0 < K; k0 += 32) {
    v8bf alo = *(const v8bf*)(Arow + k0);
    v8bf ahi = *(const v8bf*)(Arow + k0 + 8);
    v16bf a = __builtin_shufflevector(alo, ahi, 0,1,2,3,4,5,6,7,8,9,10,11,12,13,14,15);
#pragma unroll
    for (int nt = 0; nt < 4; ++nt) {
      v16bf b = *(const v16bf*)(B0 + (long)nt * 16 * K + k0);
      acc[nt] = __builtin_amdgcn_wmma_f32_16x16x32_bf16(
          /*neg_a=*/false, a, /*neg_b=*/false, b,
          /*c_mod=*/(short)0, acc[nt], /*reuse_a=*/false, /*reuse_b=*/false);
    }
  }

  __syncthreads();  // all A reads in this block done before in-place store

#pragma unroll
  for (int nt = 0; nt < 4; ++nt) {
    int c = col0 + nt * 16 + m;
    float bv = bias[c];
#pragma unroll
    for (int v = 0; v < 8; ++v) {
      float val = acc[nt][v] + bv;
      if (relu) val = (val > 0.f) ? val : 0.f;
      C[(long)(row0 + 8 * hv + v) * ldc + c] = (__bf16)val;
    }
  }
}

// final head: out[i, j] = sum_k x[i,k] * oW[k,j] + ob[j]   (10000 x 8)
__global__ void k_out(const __bf16* __restrict__ X, const float* __restrict__ oW,
                      const float* __restrict__ ob, float* __restrict__ out) {
  int t = blockIdx.x * blockDim.x + threadIdx.x;
  if (t < N_NEWS * OUTD) {
    int i = t >> 3, j = t & 7;
    float s = ob[j];
    const __bf16* xp = X + (long)i * KCAT;
#pragma unroll 8
    for (int k = 0; k < HD; ++k) s += (float)xp[k] * oW[k * OUTD + j];
    out[t] = s;
  }
}

// ---------------- orchestration ----------------

extern "C" void kernel_launch(void* const* d_in, const int* in_sizes, int n_in,
                              void* d_out, int out_size, void* d_ws, size_t ws_size,
                              hipStream_t stream) {
  const float* x_news     = (const float*)d_in[0];
  const float* x_int      = (const float*)d_in[1];
  const float* lin_news_W = (const float*)d_in[2];
  const float* lin_news_b = (const float*)d_in[3];
  const float* lin_int_W  = (const float*)d_in[4];
  const float* lin_int_b  = (const float*)d_in[5];
  const float* W_rel      = (const float*)d_in[6];   // [2,3,256,256]
  const float* W_root     = (const float*)d_in[7];   // [2,256,256]
  const float* conv_b     = (const float*)d_in[8];   // [2,256]
  const float* out_W      = (const float*)d_in[9];   // [256,8]
  const float* out_b      = (const float*)d_in[10];  // [8]
  const int* es[3] = {(const int*)d_in[11], (const int*)d_in[13], (const int*)d_in[15]};
  const int* ed[3] = {(const int*)d_in[12], (const int*)d_in[14], (const int*)d_in[16]};
  const int  nE[3] = {in_sizes[11], in_sizes[13], in_sizes[15]};
  const int soff[3] = {0, 0, N_NEWS};
  const int doff[3] = {0, N_NEWS, 0};

  // workspace carve-up (256B aligned)
  char* w = (char*)d_ws;
  auto carve = [&](size_t bytes) { void* p = (void*)w; w += (bytes + 255) & ~(size_t)255; return p; };
  __bf16* Ain  = (__bf16*)carve((size_t)NN * F_IN * 2);      // encode staging (bf16 inputs)
  __bf16* Abuf = (__bf16*)carve((size_t)NN * KCAT * 2);      // [x | agg0 | agg1 | agg2] bf16
  float*  agg  = (float*) carve((size_t)NN * HD * 4);        // one relation at a time
  float*  inv  = (float*) carve((size_t)3 * NN * 4);         // inv-degree per relation
  __bf16* BtN  = (__bf16*)carve((size_t)HD * F_IN * 2);      // lin_news_W^T
  __bf16* BtI  = (__bf16*)carve((size_t)HD * F_IN * 2);      // lin_int_W^T
  __bf16* BtL  = (__bf16*)carve((size_t)2 * HD * KCAT * 2);  // per-layer [W_root;W_rel*]^T

  const int TB = 256;
  auto gs = [](long n) { return dim3((unsigned)((n + 255) / 256)); };

  // 1) per-relation inverse in-degree
  k_zero<<<gs(3L * NN), TB, 0, stream>>>(inv, 3L * NN);
  for (int r = 0; r < 3; ++r)
    k_deg<<<gs(nE[r]), TB, 0, stream>>>(ed[r], nE[r], doff[r], inv + (long)r * NN);
  k_inv<<<gs(3L * NN), TB, 0, stream>>>(inv, 3L * NN);

  // 2) bf16 conversions: inputs + packed/transposed weights
  k_cvt<<<gs((long)N_NEWS * F_IN), TB, 0, stream>>>(x_news, Ain, (long)N_NEWS * F_IN);
  k_cvt<<<gs((long)N_INT * F_IN), TB, 0, stream>>>(x_int, Ain + (long)N_NEWS * F_IN, (long)N_INT * F_IN);
  k_packBt<<<gs((long)F_IN * HD), TB, 0, stream>>>(lin_news_W, F_IN, BtN, F_IN, 0);
  k_packBt<<<gs((long)F_IN * HD), TB, 0, stream>>>(lin_int_W,  F_IN, BtI, F_IN, 0);
  for (int l = 0; l < 2; ++l) {
    __bf16* B = BtL + (long)l * HD * KCAT;
    k_packBt<<<gs((long)HD * HD), TB, 0, stream>>>(W_root + (long)l * HD * HD, HD, B, KCAT, 0);
    for (int r = 0; r < 3; ++r)
      k_packBt<<<gs((long)HD * HD), TB, 0, stream>>>(W_rel + (long)(l * 3 + r) * HD * HD, HD, B, KCAT, HD + r * HD);
  }

  // 3) encode GEMMs -> Abuf cols [0,256)
  k_gemm<<<dim3(N_NEWS / 16), 128, 0, stream>>>(Ain, F_IN, BtN, F_IN, lin_news_b, Abuf, KCAT, 0);
  k_gemm<<<dim3(N_INT / 16), 128, 0, stream>>>(Ain + (long)N_NEWS * F_IN, F_IN, BtI, F_IN,
                                               lin_int_b, Abuf + (long)N_NEWS * KCAT, KCAT, 0);

  // 4) RGCN layers: scatter-aggregate per relation, then one fused WMMA GEMM
  for (int l = 0; l < 2; ++l) {
    for (int r = 0; r < 3; ++r) {
      k_zero<<<gs((long)NN * HD), TB, 0, stream>>>(agg, (long)NN * HD);
      k_scatter<<<gs((long)nE[r] * 64), TB, 0, stream>>>(es[r], ed[r], nE[r], soff[r], doff[r], Abuf, agg);
      k_scale_cvt<<<gs((long)NN * HD), TB, 0, stream>>>(agg, inv + (long)r * NN, Abuf, HD + r * HD);
    }
    k_gemm<<<dim3(NN / 16), 128, 0, stream>>>(Abuf, KCAT, BtL + (long)l * HD * KCAT, KCAT,
                                              conv_b + (long)l * HD, Abuf, KCAT, 1);
  }

  // 5) output head
  k_out<<<gs((long)N_NEWS * OUTD), TB, 0, stream>>>(Abuf, out_W, out_b, (float*)d_out);
}
